// STICKYLlamaAttention_39694087749940
// MI455X (gfx1250) — matrix operand order repaired and checked by reading, not compiled, not run on hardware
//
#include <hip/hip_runtime.h>

#define S_LEN 2048
#define SP    (S_LEN + 32)   // padded key stride for transposed V
#define HID_  2048
#define NH    16
#define NKV   4
#define DH    128
#define GQ    (NH / NKV)

typedef __attribute__((ext_vector_type(16))) __bf16 v16bf;
typedef __attribute__((ext_vector_type(8)))  float  v8f;
typedef __attribute__((ext_vector_type(4)))  unsigned int u32x4;
typedef __attribute__((ext_vector_type(8)))  int i32x8;
typedef __attribute__((ext_vector_type(4)))  int i32x4;

__device__ __forceinline__ v8f wmma_bf16(v16bf a, v16bf b, v8f c) {
  return __builtin_amdgcn_wmma_f32_16x16x32_bf16(false, a, false, b, (short)0, c,
                                                 false, false);
}

// ---- Tensor Data Mover: DMA a rows x 128-col bf16 tile (row stride 128) ----
// 2D descriptor: data_size=2B, tensor_dim0=128, dim0_stride=128,
// tensor_dim1=S_LEN (rows beyond return zero), tile = 128 x 32.
__device__ __forceinline__ void tdm_load_k_tile(const __bf16* gsrc,
                                                unsigned lds_off) {
  unsigned long long ga = (unsigned long long)(size_t)gsrc;
  u32x4 g0;
  g0[0] = 1u;                                  // count=1 (valid user D#)
  g0[1] = lds_off;                             // lds_addr (bytes)
  g0[2] = (unsigned)(ga & 0xFFFFFFFFu);        // global_addr[31:0]
  g0[3] = (unsigned)((ga >> 32) & 0x01FFFFFFu) // global_addr[56:32]
          | 0x80000000u;                       // type=2 ("image")
  i32x8 g1;
  g1[0] = 0x00010000;            // data_size=1 (2 bytes); wg_mask=0
  g1[1] = (int)(DH << 16);       // tensor_dim0[15:0] = 128 at bits 63:48
  g1[2] = (int)((S_LEN & 0xFFFF) << 16);  // tensor_dim1[15:0] at bits 95:80
  g1[3] = (int)(DH << 16);       // tile_dim0 = 128 at bits 127:112
  g1[4] = 32;                    // tile_dim1 = 32 rows
  g1[5] = DH;                    // tensor_dim0_stride[31:0] = 128
  g1[6] = 0;
  g1[7] = 0;
  i32x4 g2 = {0, 0, 0, 0};
  i32x4 g3 = {0, 0, 0, 0};
  i32x8 g4 = {0, 0, 0, 0, 0, 0, 0, 0};
  __builtin_amdgcn_tensor_load_to_lds(g0, g1, g2, g3, g4, 0);
}

// ---------------- elementwise helpers ----------------

__global__ void k_f2bf(const float* __restrict__ src, __bf16* __restrict__ dst, int n) {
  int i = blockIdx.x * blockDim.x + threadIdx.x;
  if (i < n) dst[i] = (__bf16)src[i];
}

__global__ void k_zero(float* __restrict__ p, int n) {
  int i = blockIdx.x * blockDim.x + threadIdx.x;
  if (i < n) p[i] = 0.f;
}

__global__ void k_fill_bf(__bf16* __restrict__ p, int n) {
  int i = blockIdx.x * blockDim.x + threadIdx.x;
  if (i < n) p[i] = (__bf16)0.f;
}

// RoPE + repack [s, nh*D] fp32 -> [nh, s, D] bf16 (for Q and K)
__global__ void k_rope_pack(const float* __restrict__ src, __bf16* __restrict__ dst,
                            int nh) {
  long long idx = (long long)blockIdx.x * blockDim.x + threadIdx.x;
  long long total = (long long)S_LEN * nh * DH;
  if (idx >= total) return;
  int d  = (int)(idx % DH);
  int hh = (int)((idx / DH) % nh);
  int s  = (int)(idx / ((long long)DH * nh));
  size_t base = (size_t)s * nh * DH + (size_t)hh * DH;
  float x = src[base + d];
  int j = d & 63;
  float inv = powf(10000.f, -(float)(2 * j) * (1.f / 128.f));
  float ang = (float)s * inv;
  float c = cosf(ang), sn = sinf(ang);
  float other = (d < 64) ? -src[base + d + 64] : src[base + d - 64];
  dst[((size_t)hh * S_LEN + s) * DH + d] = (__bf16)(x * c + other * sn);
}

// Pack V transposed: [s, hkv*D] fp32 -> Vt[hkv, d, s] bf16 (stride SP, padded)
__global__ void k_pack_vt(const float* __restrict__ src, __bf16* __restrict__ dst) {
  long long idx = (long long)blockIdx.x * blockDim.x + threadIdx.x;
  long long total = (long long)S_LEN * NKV * DH;
  if (idx >= total) return;
  int d  = (int)(idx % DH);
  int hh = (int)((idx / DH) % NKV);
  int s  = (int)(idx / ((long long)DH * NKV));
  dst[((size_t)hh * DH + d) * SP + s] =
      (__bf16)src[(size_t)s * NKV * DH + (size_t)hh * DH + d];
}

// ------- bf16 WMMA GEMM: C[M,N] = A[M,K] * B[N,K]^T, 32x32 tile / wave -------
__global__ __launch_bounds__(32) void k_gemm_bf16(
    const __bf16* __restrict__ A, int lda,
    const __bf16* __restrict__ B, int ldb,
    float* __restrict__ C, int ldc, int K) {
  const int lane = threadIdx.x;
  const int nl = lane & 15, half = lane >> 4;
  const int n0 = blockIdx.x * 32, m0 = blockIdx.y * 32;
  const __bf16* Ar[2] = {A + (size_t)(m0 + nl) * lda,
                         A + (size_t)(m0 + 16 + nl) * lda};
  const __bf16* Br[2] = {B + (size_t)(n0 + nl) * ldb,
                         B + (size_t)(n0 + 16 + nl) * ldb};
  v8f acc[2][2] = {};
  for (int k = 0; k < K; k += 32) {
    v16bf a[2], b[2];
#pragma unroll
    for (int t = 0; t < 2; ++t) {
#pragma unroll
      for (int e = 0; e < 8; ++e) {
        a[t][e]     = Ar[t][k + half * 16 + e];
        a[t][e + 8] = Ar[t][k + half * 16 + 8 + e];
      }
#pragma unroll
      for (int e = 0; e < 16; ++e) b[t][e] = Br[t][k + half * 16 + e];
      __builtin_prefetch(Ar[t] + k + 256, 0, 1);
      __builtin_prefetch(Br[t] + k + 256, 0, 1);
    }
#pragma unroll
    for (int mi = 0; mi < 2; ++mi)
#pragma unroll
      for (int ni = 0; ni < 2; ++ni)
        acc[mi][ni] = wmma_bf16(a[mi], b[ni], acc[mi][ni]);
  }
#pragma unroll
  for (int mi = 0; mi < 2; ++mi)
#pragma unroll
    for (int ni = 0; ni < 2; ++ni)
#pragma unroll
      for (int i = 0; i < 8; ++i)
        C[(size_t)(m0 + mi * 16 + i + 8 * half) * ldc + n0 + ni * 16 + nl] =
            acc[mi][ni][i];
}

// ---------------- attention pass 1: per-row logsumexp ----------------
__global__ __launch_bounds__(32) void k_attn_lse(
    const __bf16* __restrict__ Qb, const __bf16* __restrict__ Kb,
    float* __restrict__ lse) {
  const int h = blockIdx.y, q0 = blockIdx.x * 16, hkv = h / GQ;
  const int lane = threadIdx.x, nl = lane & 15, half = lane >> 4;
  const float scale = 0.08838834764831845f;  // 1/sqrt(128)
  const __bf16* Qr = Qb + ((size_t)h * S_LEN + q0 + nl) * DH;
  float m_i[8], l_i[8];
#pragma unroll
  for (int i = 0; i < 8; ++i) { m_i[i] = -1e30f; l_i[i] = 0.f; }
  for (int t0 = 0; t0 <= q0; t0 += 16) {
    const __bf16* Kr = Kb + ((size_t)hkv * S_LEN + t0 + nl) * DH;
    v8f acc = {};
#pragma unroll
    for (int kc = 0; kc < 4; ++kc) {
      v16bf a, b;
      const int kb = kc * 32;
#pragma unroll
      for (int e = 0; e < 8; ++e) {
        a[e]     = Qr[kb + half * 16 + e];
        a[e + 8] = Qr[kb + half * 16 + 8 + e];
      }
#pragma unroll
      for (int e = 0; e < 16; ++e) b[e] = Kr[kb + half * 16 + e];
      acc = wmma_bf16(a, b, acc);
    }
#pragma unroll
    for (int i = 0; i < 8; ++i) {
      const int m = i + 8 * half;
      float sv = acc[i] * scale;
      if (t0 + nl > q0 + m) sv = -1e30f;  // causal mask
      float tm = sv;
#pragma unroll
      for (int msk = 1; msk < 16; msk <<= 1)
        tm = fmaxf(tm, __shfl_xor(tm, msk, 32));
      float mn = fmaxf(m_i[i], tm);
      float p = __expf(sv - mn);
#pragma unroll
      for (int msk = 1; msk < 16; msk <<= 1) p += __shfl_xor(p, msk, 32);
      l_i[i] = l_i[i] * __expf(m_i[i] - mn) + p;
      m_i[i] = mn;
    }
  }
  if (nl == 0) {
#pragma unroll
    for (int i = 0; i < 8; ++i)
      lse[(size_t)h * S_LEN + q0 + i + 8 * half] = m_i[i] + __logf(l_i[i]);
  }
}

// ------- attention pass 2: O = softmax(QK^T) V, + per-key score mass --------
// K tiles DMA'd to LDS by the Tensor Data Mover, double-buffered.
__global__ __launch_bounds__(32) void k_attn_pv(
    const __bf16* __restrict__ Qb, const __bf16* __restrict__ Kb,
    const __bf16* __restrict__ Vt, const float* __restrict__ lse,
    __bf16* __restrict__ attn, float* __restrict__ scores) {
  __shared__ float  Pt[16][33];      // probability tile (16 q x 32 keys)
  __shared__ __bf16 Kt[2][32 * DH];  // double-buffered K tiles (2 x 8KB)
  const int h = blockIdx.y, q0 = blockIdx.x * 16, hkv = h / GQ;
  const int lane = threadIdx.x, nl = lane & 15, half = lane >> 4;
  const float scale = 0.08838834764831845f;
  const __bf16* Qr = Qb + ((size_t)h * S_LEN + q0 + nl) * DH;
  const __bf16* Kbase = Kb + (size_t)hkv * S_LEN * DH;
  float lse_v[8];
#pragma unroll
  for (int i = 0; i < 8; ++i)
    lse_v[i] = lse[(size_t)h * S_LEN + q0 + i + 8 * half];
  v8f o[8];
#pragma unroll
  for (int nd = 0; nd < 8; ++nd) o[nd] = (v8f){};

  // Prime first K tile via TDM.
  tdm_load_k_tile(Kbase, (unsigned)(size_t)&Kt[0][0]);

  for (int t0 = 0; t0 <= q0; t0 += 32) {
    const int buf = (t0 >> 5) & 1;
    const bool more = (t0 + 32) <= q0;
    if (more)  // kick next tile while current one computes
      tdm_load_k_tile(Kbase + (size_t)(t0 + 32) * DH,
                      (unsigned)(size_t)&Kt[buf ^ 1][0]);
    if (more) __builtin_amdgcn_s_wait_tensorcnt(1);
    else      __builtin_amdgcn_s_wait_tensorcnt(0);
    __syncthreads();

#pragma unroll
    for (int j = 0; j < 2; ++j) {
      const int trow = t0 + j * 16 + nl;
      const __bf16* Kr = &Kt[buf][(size_t)(j * 16 + nl) * DH];
      v8f acc = {};
#pragma unroll
      for (int kc = 0; kc < 4; ++kc) {
        v16bf a, b;
        const int kb = kc * 32;
#pragma unroll
        for (int e = 0; e < 8; ++e) {
          a[e]     = Qr[kb + half * 16 + e];
          a[e + 8] = Qr[kb + half * 16 + 8 + e];
        }
#pragma unroll
        for (int e = 0; e < 16; ++e) b[e] = Kr[kb + half * 16 + e];
        acc = wmma_bf16(a, b, acc);
      }
      float csum = 0.f;
#pragma unroll
      for (int i = 0; i < 8; ++i) {
        const int m = i + 8 * half;
        float sv = acc[i] * scale;
        if (trow > q0 + m || trow >= S_LEN) sv = -1e30f;
        const float w = __expf(sv - lse_v[i]);  // exactly-normalized prob
        Pt[m][j * 16 + nl] = w;
        csum += w;
      }
      csum += __shfl_xor(csum, 16, 32);  // combine row halves (M 0-7 & 8-15)
      if (half == 0 && trow < S_LEN && trow <= q0 + 15)
        atomicAdd(&scores[(size_t)hkv * S_LEN + trow], csum);
    }
    __syncthreads();
    // Reload P as WMMA A-matrix (16x32 bf16)
    v16bf pa;
#pragma unroll
    for (int e = 0; e < 8; ++e) {
      pa[e]     = (__bf16)Pt[nl][half * 16 + e];
      pa[e + 8] = (__bf16)Pt[nl][half * 16 + 8 + e];
    }
    __syncthreads();
    // O += P * V: V is pre-transposed so each lane reads 32 contiguous bytes.
#pragma unroll
    for (int nd = 0; nd < 8; ++nd) {
      const __bf16* vr =
          Vt + ((size_t)hkv * DH + nd * 16 + nl) * SP + t0 + half * 16;
      v16bf vb;
#pragma unroll
      for (int e = 0; e < 16; ++e) vb[e] = vr[e];
      o[nd] = wmma_bf16(pa, vb, o[nd]);
    }
  }
#pragma unroll
  for (int nd = 0; nd < 8; ++nd)
#pragma unroll
    for (int i = 0; i < 8; ++i)
      attn[(size_t)(q0 + i + 8 * half) * (NH * DH) + h * DH + nd * 16 + nl] =
          (__bf16)o[nd][i];
}

// ---------------- launcher ----------------

extern "C" void kernel_launch(void* const* d_in, const int* in_sizes, int n_in,
                              void* d_out, int out_size, void* d_ws, size_t ws_size,
                              hipStream_t stream) {
  const float* hid = (const float*)d_in[0];
  const float* Wq  = (const float*)d_in[1];
  const float* Wk  = (const float*)d_in[2];
  const float* Wv  = (const float*)d_in[3];
  const float* Wo  = (const float*)d_in[4];
  float* out    = (float*)d_out;
  float* scores = out + (size_t)S_LEN * HID_;  // [NKV, S] appended

  char* ws = (char*)d_ws;
  size_t off = 0;
  auto alloc = [&](size_t bytes) -> void* {
    void* p = ws + off;
    off += (bytes + 255) & ~(size_t)255;
    return p;
  };
  __bf16* hid_bf = (__bf16*)alloc((size_t)S_LEN * HID_ * 2);
  __bf16* Wq_bf  = (__bf16*)alloc((size_t)NH * DH * HID_ * 2);
  __bf16* Wk_bf  = (__bf16*)alloc((size_t)NKV * DH * HID_ * 2);
  __bf16* Wv_bf  = (__bf16*)alloc((size_t)NKV * DH * HID_ * 2);
  __bf16* Wo_bf  = (__bf16*)alloc((size_t)HID_ * NH * DH * 2);
  float*  Qf     = (float*)alloc((size_t)S_LEN * NH * DH * 4);
  float*  Kf     = (float*)alloc((size_t)S_LEN * NKV * DH * 4);
  float*  Vf     = (float*)alloc((size_t)S_LEN * NKV * DH * 4);
  __bf16* Qb     = (__bf16*)alloc((size_t)NH * S_LEN * DH * 2);
  __bf16* Kb     = (__bf16*)alloc((size_t)NKV * S_LEN * DH * 2);
  __bf16* Vt     = (__bf16*)alloc((size_t)NKV * DH * SP * 2);
  float*  lse    = (float*)alloc((size_t)NH * S_LEN * 4);
  __bf16* attn_bf= (__bf16*)alloc((size_t)S_LEN * NH * DH * 2);

  auto cvt = [&](const float* s, __bf16* d, int n) {
    k_f2bf<<<(n + 255) / 256, 256, 0, stream>>>(s, d, n);
  };
  cvt(hid, hid_bf, S_LEN * HID_);
  cvt(Wq,  Wq_bf,  NH * DH * HID_);
  cvt(Wk,  Wk_bf,  NKV * DH * HID_);
  cvt(Wv,  Wv_bf,  NKV * DH * HID_);
  cvt(Wo,  Wo_bf,  HID_ * NH * DH);

  dim3 blk(32);
  // QKV projections (32x32 tile per wave)
  k_gemm_bf16<<<dim3((NH * DH) / 32, S_LEN / 32), blk, 0, stream>>>(
      hid_bf, HID_, Wq_bf, HID_, Qf, NH * DH, HID_);
  k_gemm_bf16<<<dim3((NKV * DH) / 32, S_LEN / 32), blk, 0, stream>>>(
      hid_bf, HID_, Wk_bf, HID_, Kf, NKV * DH, HID_);
  k_gemm_bf16<<<dim3((NKV * DH) / 32, S_LEN / 32), blk, 0, stream>>>(
      hid_bf, HID_, Wv_bf, HID_, Vf, NKV * DH, HID_);

  // RoPE + repack Q/K; zero-padded transpose for V
  {
    long long nq = (long long)S_LEN * NH * DH;
    long long nk = (long long)S_LEN * NKV * DH;
    k_rope_pack<<<(int)((nq + 255) / 256), 256, 0, stream>>>(Qf, Qb, NH);
    k_rope_pack<<<(int)((nk + 255) / 256), 256, 0, stream>>>(Kf, Kb, NKV);
    int nvt = NKV * DH * SP;
    k_fill_bf<<<(nvt + 255) / 256, 256, 0, stream>>>(Vt, nvt);
    k_pack_vt<<<(int)((nk + 255) / 256), 256, 0, stream>>>(Vf, Vt);
  }

  // Flash attention (two-pass: exact scores accumulation)
  k_attn_lse<<<dim3(S_LEN / 16, NH), blk, 0, stream>>>(Qb, Kb, lse);
  k_zero<<<(NKV * S_LEN + 255) / 256, 256, 0, stream>>>(scores, NKV * S_LEN);
  k_attn_pv<<<dim3(S_LEN / 16, NH), blk, 0, stream>>>(Qb, Kb, Vt, lse, attn_bf,
                                                      scores);

  // Output projection -> d_out
  k_gemm_bf16<<<dim3(HID_ / 32, S_LEN / 32), blk, 0, stream>>>(
      attn_bf, NH * DH, Wo_bf, NH * DH, out, HID_, NH * DH);

  (void)in_sizes; (void)n_in; (void)out_size; (void)ws_size;
}
